// Lookup_8710193676603
// MI455X (gfx1250) — matrix operand order repaired and checked by reading, not compiled, not run on hardware
//
#include <hip/hip_runtime.h>
#include <hip/hip_bf16.h>

// ---------------------------------------------------------------------------
// RAFT correlation pyramid + lookup for MI455X (gfx1250, wave32, WMMA).
//   B=4, C=256, H8=W8=64 -> P=4096 positions per image.
//   corr[b,ij,uv] = sum_c f1[b,c,ij]*f2[b,c,uv] / 16
//   pyramid: 2x2 avg pool over (u,v), 3 times.  lookup: 41 diamond offsets,
//   bilinear, zero pad, coord scale (h-1)/h.  out: [B,64,64,4,41] f32.
// ---------------------------------------------------------------------------

#define BATCH 4
#define CDIM  256
#define P     4096      // 64*64
#define NOFF  41
#define OUT_N (BATCH * P * 4 * NOFF)   // 2,686,976

typedef __attribute__((ext_vector_type(16))) __bf16 v16bf;
typedef __attribute__((ext_vector_type(8)))  __bf16 v8bf;
typedef __attribute__((ext_vector_type(8)))  float  v8f;

union ABu { v16bf v; v8bf h[2]; };

// ---------------------------------------------------------------------------
// Kernel 1: fp32 [B,C,P] -> transposed bf16 hi/lo planes [B,P,C].
// LDS 32x33 tile transpose: coalesced reads and writes.
// ---------------------------------------------------------------------------
__global__ __launch_bounds__(256) void split_bf16_tr(
    const float* __restrict__ in, __bf16* __restrict__ hi, __bf16* __restrict__ lo)
{
    __shared__ float t[32][33];
    const int b  = blockIdx.z;
    const int c0 = blockIdx.y * 32;
    const int p0 = blockIdx.x * 32;
    const int tx = threadIdx.x;          // 0..31
    const int ty = threadIdx.y;          // 0..7
    #pragma unroll
    for (int r = 0; r < 32; r += 8)
        t[ty + r][tx] = in[((size_t)(b * CDIM + c0 + ty + r)) * P + p0 + tx];
    __syncthreads();
    #pragma unroll
    for (int r = 0; r < 32; r += 8) {
        float  f = t[tx][ty + r];
        __bf16 h = (__bf16)f;
        __bf16 l = (__bf16)(f - (float)h);
        size_t o = ((size_t)b * P + p0 + ty + r) * CDIM + c0 + tx;
        hi[o] = h;
        lo[o] = l;
    }
}

// ---------------------------------------------------------------------------
// Kernel 2: batched GEMM (WMMA bf16 split) + fused 3-level pyramid pooling.
// Grid: (256 M-tiles, 8 u-slabs, B).  Block: 256 threads = 8 waves.
// Each WG: 16 (ij rows) x 512 (uv, contiguous u-slab of 8 rows x 64 cols).
// Wave w owns N sub-range [w*64, w*64+64) = four 16x16 accumulators.
// All inner-loop loads are loop-invariant base + compile-time immediate
// offset, so the K-loop is pure  global_load_b128 / v_wmma  traffic.
// ---------------------------------------------------------------------------
__device__ inline v16bf load_afrag(const __bf16* p)
{
    // A 16x32 bf16: lanes 0-15 row M, K {0-7,16-23}; lanes 16-31 K {8-15,24-31}.
    ABu u;
    u.h[0] = *(const v8bf*)(p);
    u.h[1] = *(const v8bf*)(p + 16);
    return u.v;
}

__global__ __launch_bounds__(256) void corr_gemm_pyr(
    const __bf16* __restrict__ f1h, const __bf16* __restrict__ f1l,
    const __bf16* __restrict__ f2h, const __bf16* __restrict__ f2l,
    float* __restrict__ L0, float* __restrict__ L1,
    float* __restrict__ L2, float* __restrict__ L3)
{
    __shared__ float tile[16][512];       // 32 KB

    const int b    = blockIdx.z;
    const int m0   = blockIdx.x * 16;     // ij tile base
    const int s    = blockIdx.y;          // u-slab (u in [8s, 8s+8))
    const int n0   = s * 512;             // uv base (contiguous)
    const int wv   = threadIdx.x >> 5;
    const int ln   = threadIdx.x & 31;
    const int half = ln >> 4;
    const int l16  = ln & 15;

    const size_t fb = (size_t)b * P * CDIM;

    // A: row m0+l16, K chunk offset 0 or 8 by lane half.
    const __bf16* a_h = f1h + fb + (size_t)(m0 + l16) * CDIM + (half << 3);
    const __bf16* a_l = f1l + fb + (size_t)(m0 + l16) * CDIM + (half << 3);

    // B: column (uv), lanes 0-15 K {0-7,16-23}; lanes 16-31 K {8-15,24-31}.
    // Base pointer per wave; tiles/k-steps become immediate offsets.
    const int nw = n0 + wv * 64;
    const __bf16* bc_h = f2h + fb + (size_t)(nw + l16) * CDIM + (half << 3);
    const __bf16* bc_l = f2l + fb + (size_t)(nw + l16) * CDIM + (half << 3);

    v8f acc[4] = {};

    #pragma unroll
    for (int k0 = 0; k0 < CDIM; k0 += 32) {
        v16bf ah = load_afrag(a_h + k0);
        v16bf al = load_afrag(a_l + k0);
        #pragma unroll
        for (int t = 0; t < 4; ++t) {
            v16bf bh = load_afrag(bc_h + t * (16 * CDIM) + k0);
            v16bf bl = load_afrag(bc_l + t * (16 * CDIM) + k0);
            acc[t] = __builtin_amdgcn_wmma_f32_16x16x32_bf16(
                false, ah, false, bh, (short)0, acc[t], false, false);
            acc[t] = __builtin_amdgcn_wmma_f32_16x16x32_bf16(
                false, ah, false, bl, (short)0, acc[t], false, false);
            acc[t] = __builtin_amdgcn_wmma_f32_16x16x32_bf16(
                false, al, false, bh, (short)0, acc[t], false, false);
        }
    }

    // D layout: VGPR e -> M = e + 8*half, N = l16.  Scale by 1/sqrt(C)=1/16.
    #pragma unroll
    for (int t = 0; t < 4; ++t)
        #pragma unroll
        for (int e = 0; e < 8; ++e)
            tile[e + (half << 3)][wv * 64 + t * 16 + l16] = acc[t][e] * 0.0625f;
    __syncthreads();

    const size_t row0 = (size_t)b * P + m0;   // flat (b,ij) base

    // Level 0: 16x512 streamed out with non-temporal stores (consumed once).
    #pragma unroll
    for (int it = 0; it < 32; ++it) {
        int flat = it * 256 + threadIdx.x;
        int r = flat >> 9, c = flat & 511;
        __builtin_nontemporal_store(tile[r][c],
            L0 + (row0 + r) * (size_t)P + (size_t)(n0 + c));
    }
    // Level 1: [B,P,32,32] — 16 rows x 4 u' x 32 v' = 2048 values.
    #pragma unroll
    for (int it = 0; it < 8; ++it) {
        int flat = it * 256 + threadIdx.x;
        int r  = flat >> 7;
        int du = (flat >> 5) & 3;
        int v1 = flat & 31;
        int cb = du * 128 + 2 * v1;
        float v = 0.25f * (tile[r][cb] + tile[r][cb + 1] +
                           tile[r][cb + 64] + tile[r][cb + 65]);
        L1[((row0 + r) * 32 + (size_t)(s * 4 + du)) * 32 + v1] = v;
    }
    // Level 2: [B,P,16,16] — 16 rows x 2 u'' x 16 v'' = 512 values.
    #pragma unroll
    for (int it = 0; it < 2; ++it) {
        int flat = it * 256 + threadIdx.x;
        int r   = flat >> 5;
        int du2 = (flat >> 4) & 1;
        int v2  = flat & 15;
        float v = 0.f;
        #pragma unroll
        for (int a = 0; a < 4; ++a)
            #pragma unroll
            for (int c = 0; c < 4; ++c)
                v += tile[r][(4 * du2 + a) * 64 + 4 * v2 + c];
        L2[((row0 + r) * 16 + (size_t)(s * 2 + du2)) * 16 + v2] = v * 0.0625f;
    }
    // Level 3: [B,P,8,8] — 16 rows x 1 u''' x 8 v''' = 128 values.
    if (threadIdx.x < 128) {
        int r  = threadIdx.x >> 3;
        int v3 = threadIdx.x & 7;
        float v = 0.f;
        #pragma unroll
        for (int a = 0; a < 8; ++a)
            #pragma unroll
            for (int c = 0; c < 8; ++c)
                v += tile[r][a * 64 + 8 * v3 + c];
        L3[((row0 + r) * 8 + (size_t)s) * 8 + v3] = v * 0.015625f;
    }
}

// ---------------------------------------------------------------------------
// Kernel 3: flow-guided bilinear lookup over the 4-level pyramid.
// One thread per output element [B,64,64,4,41].
// ---------------------------------------------------------------------------
__global__ __launch_bounds__(256) void lookup_kernel(
    const float* __restrict__ flow,
    const float* __restrict__ L0, const float* __restrict__ L1,
    const float* __restrict__ L2, const float* __restrict__ L3,
    float* __restrict__ out)
{
    int gid = blockIdx.x * 256 + threadIdx.x;
    if (gid >= OUT_N) return;

    int l = gid % NOFF;
    int k = (gid / NOFF) & 3;
    int j = (gid / (NOFF * 4)) & 63;
    int i = (gid / (NOFF * 4 * 64)) & 63;
    int b =  gid / (NOFF * 4 * P);

    // diamond offset l -> (dy, dx), radius 4, row widths 1,3,5,7,9,7,5,3,1
    int rem = l, dy = -4;
    for (; dy <= 4; ++dy) {
        int cnt = 2 * (4 - abs(dy)) + 1;
        if (rem < cnt) break;
        rem -= cnt;
    }
    int dx = rem - (4 - abs(dy));

    const float* vol = (k == 0) ? L0 : (k == 1) ? L1 : (k == 2) ? L2 : L3;
    const int h = 64 >> k;

    float fdy = flow[((size_t)(b * 2 + 0) * 64 + i) * 64 + j];
    float fdx = flow[((size_t)(b * 2 + 1) * 64 + i) * 64 + j];
    float sc  = 1.0f / (float)(1 << k);
    float shr = (float)(h - 1) / (float)h;     // align_corners grid_sample factor

    float py = (((float)i + fdy) * sc + (float)dy) * shr;
    float px = (((float)j + fdx) * sc + (float)dx) * shr;

    float y0f = floorf(py), x0f = floorf(px);
    float wy1 = py - y0f, wy0 = 1.0f - wy1;
    float wx1 = px - x0f, wx0 = 1.0f - wx1;
    int y0 = (int)y0f, x0 = (int)x0f;

    size_t base = (size_t)(b * P + i * 64 + j) * h * h;
    auto g = [&](int yi, int xi) -> float {
        if (yi < 0 || yi >= h || xi < 0 || xi >= h) return 0.0f;
        return vol[base + (size_t)yi * h + xi];
    };

    out[gid] = wy0 * wx0 * g(y0, x0)     + wy0 * wx1 * g(y0, x0 + 1)
             + wy1 * wx0 * g(y0 + 1, x0) + wy1 * wx1 * g(y0 + 1, x0 + 1);
}

// ---------------------------------------------------------------------------
// Host-side launcher.  Workspace layout (bytes):
//   f1 hi | f1 lo | f2 hi | f2 lo : 4 x 8 MiB bf16 planes [B,P,C]
//   L0: 256 MiB  L1: 64 MiB  L2: 16 MiB  L3: 4 MiB   (fp32 pyramid)
// ---------------------------------------------------------------------------
extern "C" void kernel_launch(void* const* d_in, const int* in_sizes, int n_in,
                              void* d_out, int out_size, void* d_ws, size_t ws_size,
                              hipStream_t stream)
{
    const float* feat1 = (const float*)d_in[0];
    const float* feat2 = (const float*)d_in[1];
    const float* flow  = (const float*)d_in[2];
    float* out = (float*)d_out;

    const size_t PLANE = (size_t)BATCH * P * CDIM * sizeof(__bf16);   // 8 MiB
    char* ws = (char*)d_ws;
    __bf16* f1h = (__bf16*)(ws + 0 * PLANE);
    __bf16* f1l = (__bf16*)(ws + 1 * PLANE);
    __bf16* f2h = (__bf16*)(ws + 2 * PLANE);
    __bf16* f2l = (__bf16*)(ws + 3 * PLANE);
    size_t off = 4 * PLANE;
    float* L0 = (float*)(ws + off); off += (size_t)BATCH * P * 4096 * sizeof(float);
    float* L1 = (float*)(ws + off); off += (size_t)BATCH * P * 1024 * sizeof(float);
    float* L2 = (float*)(ws + off); off += (size_t)BATCH * P * 256  * sizeof(float);
    float* L3 = (float*)(ws + off); off += (size_t)BATCH * P * 64   * sizeof(float);
    if (ws_size < off) return;   // workspace too small: deterministic no-op

    // 1) bf16 hi/lo split + transpose to [B,P,C]
    dim3 cgrid(P / 32, CDIM / 32, BATCH), cblk(32, 8);
    split_bf16_tr<<<cgrid, cblk, 0, stream>>>(feat1, f1h, f1l);
    split_bf16_tr<<<cgrid, cblk, 0, stream>>>(feat2, f2h, f2l);

    // 2) WMMA GEMM + fused pyramid
    dim3 ggrid(P / 16, 8, BATCH);
    corr_gemm_pyr<<<ggrid, 256, 0, stream>>>(f1h, f1l, f2h, f2l, L0, L1, L2, L3);

    // 3) lookup
    lookup_kernel<<<(OUT_N + 255) / 256, 256, 0, stream>>>(flow, L0, L1, L2, L3, out);
}
